// ProjectedAdaptiveLogSoftmax_52192442581862
// MI455X (gfx1250) — matrix operand order repaired and checked by reading, not compiled, not run on hardware
//
#include <hip/hip_runtime.h>

typedef _Float16 half_t;
typedef __attribute__((ext_vector_type(16))) _Float16 v16h;
typedef __attribute__((ext_vector_type(8)))  _Float16 v8h;
typedef __attribute__((ext_vector_type(8)))  float    v8f;
typedef __attribute__((ext_vector_type(4)))  float    v4f;

#define NEG_BIG (-1.0e30f)

__device__ __forceinline__ v8f wmma16(v16h a, v16h b, v8f c) {
  // D(16x16 f32) = A(16x32 f16) * B(32x16 f16) + C
  return __builtin_amdgcn_wmma_f32_16x16x32_f16(false, a, false, b, (short)0, c,
                                                false, false);
}

// A fragment (this code's contiguous k order, same as B): lane(g,hi) holds halves
// [k0+16*hi, +8) in v[0..7] and [k0+16*hi+8, +8) in v[8..15] of row g.
__device__ __forceinline__ v16h load_a_frag(const half_t* rowp, int k0, int hi) {
  const v8h lo = *(const v8h*)(rowp + k0 + 16 * hi);
  const v8h up = *(const v8h*)(rowp + k0 + 16 * hi + 8);
  v16h r;
#pragma unroll
  for (int i = 0; i < 8; ++i) { r[i] = lo[i]; r[8 + i] = up[i]; }
  return r;
}

// B fragment (this code's contiguous k order, same as A): lane(g,hi) holds halves
// [k0+16*hi, +16) of the W/projT row for column g.
__device__ __forceinline__ v16h load_b_frag_h(const half_t* rowp, int k0, int hi) {
  const v8h lo = *(const v8h*)(rowp + k0 + 16 * hi);
  const v8h up = *(const v8h*)(rowp + k0 + 16 * hi + 8);
  v16h r;
#pragma unroll
  for (int i = 0; i < 8; ++i) { r[i] = lo[i]; r[8 + i] = up[i]; }
  return r;
}

// B fragment converted on the fly from an f32 W row; zero beyond dW (K padding).
__device__ __forceinline__ v16h cvt_b_frag(const float* wrow, int k0, int hi, int dW) {
  v16h b;
#pragma unroll
  for (int i = 0; i < 16; ++i) b[i] = (half_t)0.0f;
  const int off = k0 + 16 * hi;
  if (off < dW) {
    const v4f f0 = *(const v4f*)(wrow + off);
    const v4f f1 = *(const v4f*)(wrow + off + 4);
    const v4f f2 = *(const v4f*)(wrow + off + 8);
    const v4f f3 = *(const v4f*)(wrow + off + 12);
#pragma unroll
    for (int i = 0; i < 4; ++i) {
      b[i]      = (half_t)f0[i];
      b[4 + i]  = (half_t)f1[i];
      b[8 + i]  = (half_t)f2[i];
      b[12 + i] = (half_t)f3[i];
    }
  }
  return b;
}

// ---------------------------------------------------------------------------
// elementwise f32 -> f16
// ---------------------------------------------------------------------------
__global__ void cvt_f16_kernel(const float* __restrict__ src,
                               half_t* __restrict__ dst, int n) {
  int i = blockIdx.x * blockDim.x + threadIdx.x;
  if (i < n) dst[i] = (half_t)src[i];
}

// transpose + pad: src f32 [K x D] -> dst f16 [Dpad x K], rows >= D are zero
__global__ void cvt_transpose_pad_kernel(const float* __restrict__ src,
                                         half_t* __restrict__ dst,
                                         int K, int D, int Dpad) {
  int i = blockIdx.x * blockDim.x + threadIdx.x;  // i = c*K + k
  if (i >= Dpad * K) return;
  int c = i / K, k = i - c * K;
  dst[i] = (c < D) ? (half_t)src[(size_t)k * D + c] : (half_t)0.0f;
}

// row-pad convert: src f32 [V x dW] -> dst f16 [V x dP], cols >= dW are zero
__global__ void cvt_rows_pad_kernel(const float* __restrict__ src,
                                    half_t* __restrict__ dst,
                                    int dW, int dP, int n /* = V*dP */) {
  int i = blockIdx.x * blockDim.x + threadIdx.x;
  if (i >= n) return;
  int r = i / dP, c = i - r * dP;
  dst[i] = (c < dW) ? (half_t)src[(size_t)r * dW + c] : (half_t)0.0f;
}

// ---------------------------------------------------------------------------
// P(f16)[1024 x Dp] = A(f16)[1024 x K] @ Bt(f16)[Dp x K]^T
// one wave per 16x16 tile
// ---------------------------------------------------------------------------
__global__ void proj_gemm_f16_kernel(const half_t* __restrict__ A,
                                     const half_t* __restrict__ Bt,
                                     half_t* __restrict__ Out, int K, int Dp) {
  const int lane = threadIdx.x;
  const int g = lane & 15;
  const int hi = lane >> 4;
  const int row0 = blockIdx.x * 16;
  const int n0 = blockIdx.y * 16;

  const half_t* arow = A + (size_t)(row0 + g) * K;
  const half_t* brow = Bt + (size_t)(n0 + g) * K;

  v8f acc = {};
  for (int k = 0; k < K; k += 32) {
    v16h a = load_a_frag(arow, k, hi);
    v16h b = load_b_frag_h(brow, k, hi);
    acc = wmma16(a, b, acc);
  }
  half_t* out = Out + (size_t)(row0 + 8 * hi) * Dp + n0 + g;
#pragma unroll
  for (int e = 0; e < 8; ++e) out[(size_t)e * Dp] = (half_t)acc[e];
}

// ---------------------------------------------------------------------------
// Logits + streaming LSE for one cluster.
//   logit[n,v] = P(f16)[n,:dA] . f16(Wrow(v)) + bias(v)
//   B_F16 ? W rows come pre-converted/padded f16 (row stride dA)
//         : W rows are f32 (row stride dW), converted in registers.
//   One wave: 16 rows x 512-column chunk; 4 column tiles share the A fragment.
// ---------------------------------------------------------------------------
template <bool B_F16>
__global__ void logits_lse_kernel(
    const half_t* __restrict__ P, int dA,
    const float* __restrict__ W, const float* __restrict__ Wx,
    const half_t* __restrict__ Wh, const half_t* __restrict__ Wxh,
    const float* __restrict__ bias, const float* __restrict__ biasx,
    int dW, int split, int V,
    const int* __restrict__ target, int cap_lo, int cap_hi,
    float* __restrict__ tgt_out, float* __restrict__ chead,
    float* __restrict__ part_m, float* __restrict__ part_s, int chunk_base) {
  const int lane = threadIdx.x;
  const int g = lane & 15;
  const int hi = lane >> 4;
  const int row0 = blockIdx.x * 16;
  const int chunk = blockIdx.y;
  const int col_base = chunk * 512;

  int capcol[8];
#pragma unroll
  for (int e = 0; e < 8; ++e) {
    int t = target[row0 + 8 * hi + e] - cap_lo;
    capcol[e] = t < 0 ? 0 : (t > cap_hi ? cap_hi : t);
  }

  const half_t* arow = P + (size_t)(row0 + g) * dA;

  float rm[8], rs[8];
#pragma unroll
  for (int e = 0; e < 8; ++e) { rm[e] = NEG_BIG; rs[e] = 0.0f; }

  for (int tb = 0; tb < 32; tb += 4) {
    if (col_base + tb * 16 >= V) break;  // uniform across the wave

    const float* wr32[4];
    const half_t* wr16[4];
    float bv[4];
    bool cv[4];
    int vcol[4];
#pragma unroll
    for (int j = 0; j < 4; ++j) {
      int v = col_base + (tb + j) * 16 + g;
      vcol[j] = v;
      cv[j] = (v < V);
      int vc = cv[j] ? v : (V - 1);
      if constexpr (B_F16) {
        wr16[j] = (vc < split) ? (Wh + (size_t)vc * dA)
                               : (Wxh + (size_t)(vc - split) * dA);
      } else {
        wr32[j] = (vc < split) ? (W + (size_t)vc * dW)
                               : (Wx + (size_t)(vc - split) * dW);
      }
      bv[j] = cv[j] ? ((v < split) ? bias[v] : biasx[v - split]) : NEG_BIG;
    }
    {  // prefetch next tile group's W rows (global_prefetch_b8)
      int vp = col_base + (tb + 4) * 16 + g;
      if (vp < V) {
        if constexpr (B_F16) {
          const half_t* pf = (vp < split) ? (Wh + (size_t)vp * dA)
                                          : (Wxh + (size_t)(vp - split) * dA);
          __builtin_prefetch(pf, 0, 1);
        } else {
          const float* pf = (vp < split) ? (W + (size_t)vp * dW)
                                         : (Wx + (size_t)(vp - split) * dW);
          __builtin_prefetch(pf, 0, 1);
        }
      }
    }

    v8f acc[4] = {};
    for (int k = 0; k < dA; k += 32) {
      v16h a = load_a_frag(arow, k, hi);
#pragma unroll
      for (int j = 0; j < 4; ++j) {
        v16h bb;
        if constexpr (B_F16) bb = load_b_frag_h(wr16[j], k, hi);
        else                 bb = cvt_b_frag(wr32[j], k, hi, dW);
        acc[j] = wmma16(a, bb, acc[j]);
      }
    }

#pragma unroll
    for (int j = 0; j < 4; ++j) {
      const int v = vcol[j];
      const bool ok = cv[j];
      const float cb = bv[j];
#pragma unroll
      for (int e = 0; e < 8; ++e) {
        float x = ok ? acc[j][e] + cb : NEG_BIG;
        int row = row0 + 8 * hi + e;
        if (ok) {
          if (v == capcol[e]) tgt_out[row] = x;
          if (chead != nullptr && v >= split)
            chead[(size_t)(v - split) * 1024 + row] = x;
        }
        // row-wise reduce across the 16 lanes of this half (cols of the row)
        float tm = x;
        tm = fmaxf(tm, __shfl_xor(tm, 1, 32));
        tm = fmaxf(tm, __shfl_xor(tm, 2, 32));
        tm = fmaxf(tm, __shfl_xor(tm, 4, 32));
        tm = fmaxf(tm, __shfl_xor(tm, 8, 32));
        float ts = expf(x - tm);
        ts += __shfl_xor(ts, 1, 32);
        ts += __shfl_xor(ts, 2, 32);
        ts += __shfl_xor(ts, 4, 32);
        ts += __shfl_xor(ts, 8, 32);
        float nm = fmaxf(rm[e], tm);
        rs[e] = rs[e] * expf(rm[e] - nm) + ts * expf(tm - nm);
        rm[e] = nm;
      }
    }
  }

  if (g == 0) {  // lanes 0 / 16 write rows 0..7 / 8..15
    size_t base = (size_t)(chunk_base + chunk) * 1024 + row0 + 8 * hi;
#pragma unroll
    for (int e = 0; e < 8; ++e) {
      part_m[base + e] = rm[e];
      part_s[base + e] = rs[e];
    }
  }
}

// ---------------------------------------------------------------------------
// Merge chunk partials into per-row LSEs, assemble NLL, mean-reduce.
// chunk layout: head [0,40) t1 [40,80) t2 [80,393) t3 [393,526)
// ---------------------------------------------------------------------------
__global__ void finalize_kernel(const int* __restrict__ target,
                                const float* __restrict__ tgt_head,
                                const float* __restrict__ chead,
                                const float* __restrict__ tgt_tail,
                                const float* __restrict__ part_m,
                                const float* __restrict__ part_s,
                                float* __restrict__ out) {
  __shared__ float red[1024];
  const int r = threadIdx.x;
  const int t = target[r];
  const int cl = (t < 20000) ? 0 : (t < 40000) ? 1 : (t < 200000) ? 2 : 3;

  float M = NEG_BIG, S = 0.0f;
  for (int c = 0; c < 40; ++c) {
    float pm = part_m[(size_t)c * 1024 + r];
    float ps = part_s[(size_t)c * 1024 + r];
    float nm = fmaxf(M, pm);
    S = S * expf(M - nm) + ps * expf(pm - nm);
    M = nm;
  }
  float lse_head = M + logf(S);

  float nll;
  if (cl == 0) {
    nll = lse_head - tgt_head[r];
  } else {
    int base, nc;
    if (cl == 1) { base = 40;  nc = 40;  }
    else if (cl == 2) { base = 80;  nc = 313; }
    else { base = 393; nc = 133; }
    M = NEG_BIG; S = 0.0f;
    for (int c = 0; c < nc; ++c) {
      float pm = part_m[(size_t)(base + c) * 1024 + r];
      float ps = part_s[(size_t)(base + c) * 1024 + r];
      float nm = fmaxf(M, pm);
      S = S * expf(M - nm) + ps * expf(pm - nm);
      M = nm;
    }
    float lse_tail = M + logf(S);
    // reference quirk: tail cluster i uses head column 20003 - i
    float hl = chead[(size_t)(3 - cl) * 1024 + r];
    float tl = tgt_tail[(size_t)(cl - 1) * 1024 + r];
    nll = (lse_head - hl) + (lse_tail - tl);
  }

  red[r] = nll;
  __syncthreads();
  for (int s = 512; s > 0; s >>= 1) {
    if (r < s) red[r] += red[r + s];
    __syncthreads();
  }
  if (r == 0) out[0] = red[0] * (1.0f / 1024.0f);
}

// ---------------------------------------------------------------------------
extern "C" void kernel_launch(void* const* d_in, const int* in_sizes, int n_in,
                              void* d_out, int out_size, void* d_ws,
                              size_t ws_size, hipStream_t stream) {
  (void)in_sizes; (void)n_in; (void)out_size;
  const float* hidden = (const float*)d_in[0];
  const int* target   = (const int*)d_in[1];
  const float* W0     = (const float*)d_in[2];
  const float* b0     = (const float*)d_in[3];
  const float* proj0  = (const float*)d_in[4];
  const float* W1     = (const float*)d_in[5];
  const float* b1     = (const float*)d_in[6];
  const float* proj1  = (const float*)d_in[7];
  const float* W2     = (const float*)d_in[8];
  const float* b2     = (const float*)d_in[9];
  const float* proj2  = (const float*)d_in[10];
  const float* W3     = (const float*)d_in[11];
  const float* b3     = (const float*)d_in[12];
  const float* proj3  = (const float*)d_in[13];
  const float* cW     = (const float*)d_in[14];
  const float* cb     = (const float*)d_in[15];

  // workspace layout (all blocks are 16B-multiples -> alignment OK)
  char* p = (char*)d_ws;
  half_t* hidden_h = (half_t*)p;  p += (size_t)1024 * 1024 * 2;
  half_t* projT0   = (half_t*)p;  p += (size_t)1024 * 1024 * 2;
  half_t* projT1   = (half_t*)p;  p += (size_t)256 * 1024 * 2;
  half_t* projT2   = (half_t*)p;  p += (size_t)64 * 1024 * 2;
  half_t* projT3   = (half_t*)p;  p += (size_t)32 * 1024 * 2;   // padded 16->32
  half_t* P0h      = (half_t*)p;  p += (size_t)1024 * 1024 * 2;
  half_t* P1h      = (half_t*)p;  p += (size_t)1024 * 256 * 2;
  half_t* P2h      = (half_t*)p;  p += (size_t)1024 * 64 * 2;
  half_t* P3h      = (half_t*)p;  p += (size_t)1024 * 32 * 2;
  float* tgt_head  = (float*)p;   p += (size_t)1024 * 4;
  float* chead     = (float*)p;   p += (size_t)3 * 1024 * 4;
  float* tgt_tail  = (float*)p;   p += (size_t)3 * 1024 * 4;
  float* part_m    = (float*)p;   p += (size_t)526 * 1024 * 4;
  float* part_s    = (float*)p;   p += (size_t)526 * 1024 * 4;  // ~12 MB base
  // optional f16 weight mirrors (~76 MB extra)
  half_t* W0h = (half_t*)p;  p += (size_t)20000 * 1024 * 2;
  half_t* W1h = (half_t*)p;  p += (size_t)20000 * 256 * 2;
  half_t* W2h = (half_t*)p;  p += (size_t)160000 * 64 * 2;
  half_t* W3h = (half_t*)p;  p += (size_t)67735 * 32 * 2;       // padded 16->32
  half_t* cWh = (half_t*)p;  p += (size_t)3 * 1024 * 2;
  const bool big = ((size_t)(p - (char*)d_ws)) <= ws_size;

  // conversions (always)
  cvt_f16_kernel<<<(1024 * 1024 + 255) / 256, 256, 0, stream>>>(hidden, hidden_h,
                                                                1024 * 1024);
  cvt_transpose_pad_kernel<<<(1024 * 1024 + 255) / 256, 256, 0, stream>>>(
      proj0, projT0, 1024, 1024, 1024);
  cvt_transpose_pad_kernel<<<(256 * 1024 + 255) / 256, 256, 0, stream>>>(
      proj1, projT1, 1024, 256, 256);
  cvt_transpose_pad_kernel<<<(64 * 1024 + 255) / 256, 256, 0, stream>>>(
      proj2, projT2, 1024, 64, 64);
  cvt_transpose_pad_kernel<<<(32 * 1024 + 255) / 256, 256, 0, stream>>>(
      proj3, projT3, 1024, 16, 32);
  if (big) {  // f16 weight mirrors (padded rows where needed)
    cvt_rows_pad_kernel<<<(20000 * 1024 + 255) / 256, 256, 0, stream>>>(
        W0, W0h, 1024, 1024, 20000 * 1024);
    cvt_rows_pad_kernel<<<(20000 * 256 + 255) / 256, 256, 0, stream>>>(
        W1, W1h, 256, 256, 20000 * 256);
    cvt_rows_pad_kernel<<<(160000 * 64 + 255) / 256, 256, 0, stream>>>(
        W2, W2h, 64, 64, 160000 * 64);
    cvt_rows_pad_kernel<<<(67735 * 32 + 255) / 256, 256, 0, stream>>>(
        W3, W3h, 16, 32, 67735 * 32);
    cvt_rows_pad_kernel<<<(3 * 1024 + 255) / 256, 256, 0, stream>>>(
        cW, cWh, 1024, 1024, 3 * 1024);
  }

  dim3 blk(32);
  // projections (f16 WMMA, f16 output)
  proj_gemm_f16_kernel<<<dim3(64, 64), blk, 0, stream>>>(hidden_h, projT0, P0h, 1024, 1024);
  proj_gemm_f16_kernel<<<dim3(64, 16), blk, 0, stream>>>(hidden_h, projT1, P1h, 1024, 256);
  proj_gemm_f16_kernel<<<dim3(64, 4),  blk, 0, stream>>>(hidden_h, projT2, P2h, 1024, 64);
  proj_gemm_f16_kernel<<<dim3(64, 2),  blk, 0, stream>>>(hidden_h, projT3, P3h, 1024, 32);

  if (big) {
    logits_lse_kernel<true><<<dim3(64, 40), blk, 0, stream>>>(
        P0h, 1024, nullptr, nullptr, W0h, cWh, b0, cb, 1024, 20000, 20003,
        target, 0, 19999, tgt_head, chead, part_m, part_s, 0);
    logits_lse_kernel<true><<<dim3(64, 40), blk, 0, stream>>>(
        P1h, 256, nullptr, nullptr, W1h, W1h, b1, b1, 256, 20000, 20000,
        target, 20000, 19999, tgt_tail, nullptr, part_m, part_s, 40);
    logits_lse_kernel<true><<<dim3(64, 313), blk, 0, stream>>>(
        P2h, 64, nullptr, nullptr, W2h, W2h, b2, b2, 64, 160000, 160000,
        target, 40000, 159999, tgt_tail + 1024, nullptr, part_m, part_s, 80);
    logits_lse_kernel<true><<<dim3(64, 133), blk, 0, stream>>>(
        P3h, 32, nullptr, nullptr, W3h, W3h, b3, b3, 16, 67735, 67735,
        target, 200000, 67734, tgt_tail + 2048, nullptr, part_m, part_s, 393);
  } else {
    logits_lse_kernel<false><<<dim3(64, 40), blk, 0, stream>>>(
        P0h, 1024, W0, cW, nullptr, nullptr, b0, cb, 1024, 20000, 20003,
        target, 0, 19999, tgt_head, chead, part_m, part_s, 0);
    logits_lse_kernel<false><<<dim3(64, 40), blk, 0, stream>>>(
        P1h, 256, W1, W1, nullptr, nullptr, b1, b1, 256, 20000, 20000,
        target, 20000, 19999, tgt_tail, nullptr, part_m, part_s, 40);
    logits_lse_kernel<false><<<dim3(64, 313), blk, 0, stream>>>(
        P2h, 64, W2, W2, nullptr, nullptr, b2, b2, 64, 160000, 160000,
        target, 40000, 159999, tgt_tail + 1024, nullptr, part_m, part_s, 80);
    logits_lse_kernel<false><<<dim3(64, 133), blk, 0, stream>>>(
        P3h, 32, W3, W3, nullptr, nullptr, b3, b3, 16, 67735, 67735,
        target, 200000, 67734, tgt_tail + 2048, nullptr, part_m, part_s, 393);
  }

  finalize_kernel<<<1, 1024, 0, stream>>>(target, tgt_head, chead, tgt_tail,
                                          part_m, part_s, (float*)d_out);
}
